// LSA_11476152615108
// MI455X (gfx1250) — matrix operand order repaired and checked by reading, not compiled, not run on hardware
//
#include <hip/hip_runtime.h>

// Problem constants (match the JAX reference)
#define DIMC   1024   // model dim
#define SEQ    2048   // sequence length
#define NHEADS 16
#define DHEAD  64
#define BATCH  4
#define MROWS  (BATCH * SEQ)   // 8192 token rows

typedef __attribute__((ext_vector_type(16))) _Float16 v16h;
typedef __attribute__((ext_vector_type(8)))  _Float16 v8h;
typedef __attribute__((ext_vector_type(4)))  _Float16 v4h;
typedef __attribute__((ext_vector_type(8)))  float    v8f;

union V16U { v16h v; v8h h[2]; };

// D = A(16x32 f16) * B(32x16 f16) + C(16x16 f32)
__device__ __forceinline__ v8f wmma_f16(v16h a, v16h b, v8f c) {
  return __builtin_amdgcn_wmma_f32_16x16x32_f16(
      /*neg_a=*/false, a, /*neg_b=*/false, b,
      /*c_mod=*/(short)0, c, /*reuse_a=*/false, /*reuse_b=*/false);
}

// A-fragment 16x32 f16 from row-major data (ld in halves).
// lane<16 : row=lane, K {0..7, 16..23} ; lane>=16 : row=lane-16, K {8..15, 24..31}
__device__ __forceinline__ v16h load_frag_a(const _Float16* base, int ld) {
  const int lane = threadIdx.x & 31;
  const _Float16* p = base + (size_t)(lane & 15) * ld + ((lane & 16) ? 8 : 0);
  V16U u;
  u.h[0] = *(const v8h*)(p);
  u.h[1] = *(const v8h*)(p + 16);
  return u.v;
}

// B-fragment 32x16 f16. Source is stored "column-major as rows": the B-matrix
// column n is a contiguous run of 32 K-values at base + n*ld.
// lane<16 : col=lane, K {0..7, 16..23} ; lane>=16 : col=lane-16, K {8..15, 24..31}
__device__ __forceinline__ v16h load_frag_b(const _Float16* base, int ld) {
  const int lane = threadIdx.x & 31;
  const _Float16* p = base + (size_t)(lane & 15) * ld + ((lane & 16) ? 8 : 0);
  V16U u;
  u.h[0] = *(const v8h*)(p);
  u.h[1] = *(const v8h*)(p + 16);
  return u.v;
}

// ---------------------------------------------------------------- conversions
__global__ void cvt_f32_to_f16(const float4* __restrict__ src,
                               v4h* __restrict__ dst, int n4) {
  int i = blockIdx.x * blockDim.x + threadIdx.x;
  if (i < n4) {
    float4 s = src[i];
    v4h d = { (_Float16)s.x, (_Float16)s.y, (_Float16)s.z, (_Float16)s.w };
    dst[i] = d;
  }
}

// Wt[c][k] = (f16) W[k][c]  — makes every WMMA B-fragment load contiguous.
__global__ void transpose_cvt(const float* __restrict__ W,
                              _Float16* __restrict__ Wt) {
  __shared__ float tile[16][17];
  const int tx = threadIdx.x & 15, ty = threadIdx.x >> 4;
  tile[ty][tx] = W[(size_t)(blockIdx.y * 16 + ty) * DIMC + blockIdx.x * 16 + tx];
  __syncthreads();
  Wt[(size_t)(blockIdx.x * 16 + ty) * DIMC + blockIdx.y * 16 + tx] =
      (_Float16)tile[tx][ty];
}

// ---------------------------------------------------------------- GEMM
// C[M=8192, N=1024] = A[M,1024](f16, row-major) * Wt[N,1024](f16, pre-transposed) + bias
// Block tile 128x128 (8 waves), wave tile 32x64: per k-step 8 WMMAs against
// 6 fragment loads (A-fragments reused across 4 N-tiles). Unroll 2 keeps two
// k-steps of loads in flight (staggered s_wait_loadcnt).
// EPI 0: f16 out, natural [m][c]         (Q, K projections)
// EPI 1: f16 out, V-transposed [b,h,d,n] (V projection -> PV B-fragments contiguous)
// EPI 2: f32 out, natural [m][c]         (final output projection)
template <int EPI>
__global__ __launch_bounds__(256) void gemm_wmma(
    const _Float16* __restrict__ A, const _Float16* __restrict__ Bt,
    const float* __restrict__ bias, void* __restrict__ Out) {
  const int lane = threadIdx.x & 31;
  const int wave = threadIdx.x >> 5;
  const int m0 = blockIdx.x * 128 + (wave & 3) * 32;  // 4 waves along M
  const int n0 = blockIdx.y * 128 + (wave >> 2) * 64; // 2 waves along N
  v8f acc[2][4] = {};
  const _Float16* Ap = A + (size_t)m0 * DIMC;
  const _Float16* Bp = Bt + (size_t)n0 * DIMC;

#pragma unroll 2
  for (int k0 = 0; k0 < DIMC; k0 += 32) {
    if (k0 + 32 < DIMC) {  // hint next k-tile into cache (global_prefetch_b8)
      __builtin_prefetch(Ap + k0 + 32, 0, 3);
      __builtin_prefetch(Bp + k0 + 32, 0, 3);
    }
    v16h a0 = load_frag_a(Ap + k0, DIMC);
    v16h a1 = load_frag_a(Ap + (size_t)16 * DIMC + k0, DIMC);
    v16h bt[4];
#pragma unroll
    for (int t = 0; t < 4; ++t)
      bt[t] = load_frag_b(Bp + (size_t)(t * 16) * DIMC + k0, DIMC);
#pragma unroll
    for (int t = 0; t < 4; ++t) {
      acc[0][t] = wmma_f16(a0, bt[t], acc[0][t]);
      acc[1][t] = wmma_f16(a1, bt[t], acc[1][t]);
    }
  }

  const int rbase = (lane & 16) ? 8 : 0;  // C-layout: lane>=16 holds rows 8..15
  const int colL = lane & 15;
#pragma unroll
  for (int cn = 0; cn < 4; ++cn) {
    const int c = n0 + cn * 16 + colL;
    const float bia = bias[c];
#pragma unroll
    for (int cm = 0; cm < 2; ++cm) {
#pragma unroll
      for (int r = 0; r < 8; ++r) {
        const int m = m0 + cm * 16 + rbase + r;
        const float val = acc[cm][cn][r] + bia;
        if (EPI == 0) {
          ((_Float16*)Out)[(size_t)m * DIMC + c] = (_Float16)val;
        } else if (EPI == 1) {
          const int bb = m >> 11, n = m & (SEQ - 1);
          const int h = c >> 6, d = c & (DHEAD - 1);
          ((_Float16*)Out)[(size_t)((bb * NHEADS + h) * DHEAD + d) * SEQ + n] =
              (_Float16)val;
        } else {
          ((float*)Out)[(size_t)m * DIMC + c] = val;
        }
      }
    }
  }
}

// ---------------------------------------------------------------- attention
// Flash-style online softmax. 8 waves/WG; each wave owns 16 query rows of one
// (b,h), keeps Q A-fragments + 16x64 f32 output accumulator in registers, and
// streams K/V in 32-column blocks. V fragments are issued at the TOP of the
// block; K fragments are DOUBLE-BUFFERED across j-blocks so next-block K
// latency hides under this block's softmax + PV work.
__global__ __launch_bounds__(256) void attn_kernel(
    const _Float16* __restrict__ Qh, const _Float16* __restrict__ Kh,
    const _Float16* __restrict__ Vt, const float* __restrict__ logt,
    _Float16* __restrict__ Oh) {
  __shared__ float    sS[8][16 * 33];  // score tile, padded stride (bank-safe)
  __shared__ _Float16 sP[8][16 * 32];  // probabilities, A-fragment friendly
  __shared__ float    sF[8][16];       // per-row rescale factor broadcast

  const int lane = threadIdx.x & 31;
  const int wave = threadIdx.x >> 5;
  const int h = blockIdx.y, b = blockIdx.z;
  const int q0 = blockIdx.x * 128 + wave * 16;  // first query row of this wave
  const float temp = __expf(logt[0]);

  float* mySS = sS[wave];
  _Float16* myP = sP[wave];
  float* myF = sF[wave];

  // Q fragments (d 0..31 and 32..63) live in registers for the whole pass.
  const _Float16* qbase = Qh + ((size_t)(b * SEQ) + q0) * DIMC + h * DHEAD;
  const v16h qa0 = load_frag_a(qbase, DIMC);
  const v16h qa1 = load_frag_a(qbase + 32, DIMC);

  v8f o[4] = {};                       // 16 rows x 64 cols output accumulator
  float m_run = -__builtin_inff();     // row stats owned by lanes 0..15
  float l_run = 0.0f;
  const int rbase = (lane & 16) ? 8 : 0;
  const int colL = lane & 15;

  const _Float16* Kbh = Kh + (size_t)(b * SEQ) * DIMC + h * DHEAD;
  const _Float16* Vbh = Vt + (size_t)((b * NHEADS + h) * DHEAD) * SEQ;

  // Prime the K double-buffer: kf[0..1] = tile j0..+15 (d 0..31 | 32..63),
  // kf[2..3] = tile j0+16..+31.
  v16h kf[4];
#pragma unroll
  for (int t = 0; t < 4; ++t)
    kf[t] = load_frag_b(Kbh + (size_t)((t >> 1) * 16) * DIMC + (t & 1) * 32,
                        DIMC);

  for (int j0 = 0; j0 < SEQ; j0 += 32) {
    const _Float16* vb = Vbh + j0;

    // ---- issue V fragments early; consumed only after softmax
    v16h vf[4];
#pragma unroll
    for (int t = 0; t < 4; ++t)
      vf[t] = load_frag_b(vb + (size_t)(t * 16) * SEQ, SEQ);

    // ---- scores: S[16 q x 32 k] via 4 WMMAs (K rows are already B-layout)
    v8f s0 = {}, s1 = {};
    s0 = wmma_f16(qa0, kf[0], s0);
    s0 = wmma_f16(qa1, kf[1], s0);
    s1 = wmma_f16(qa0, kf[2], s1);
    s1 = wmma_f16(qa1, kf[3], s1);

    // ---- kick off NEXT block's K fragments; they land during softmax + PV
    if (j0 + 32 < SEQ) {
      const _Float16* kn = Kbh + (size_t)(j0 + 32) * DIMC;
#pragma unroll
      for (int t = 0; t < 4; ++t)
        kf[t] = load_frag_b(kn + (size_t)((t >> 1) * 16) * DIMC + (t & 1) * 32,
                            DIMC);
    }

    // ---- spill C-layout scores to this wave's LDS slice
#pragma unroll
    for (int r = 0; r < 8; ++r) {
      mySS[(rbase + r) * 33 + colL] = s0[r];
      mySS[(rbase + r) * 33 + 16 + colL] = s1[r];
    }
    asm volatile("s_wait_dscnt 0" ::: "memory");  // in-order LDS, fence cross-lane

    // ---- online softmax update; lane l owns row l
    if (lane < 16) {
      const int row = lane;
      const int gi = q0 + row;
      float mx = m_run;
      float sv[32];
#pragma unroll
      for (int c = 0; c < 32; ++c) {
        float v = mySS[row * 33 + c] * temp;
        if (j0 + c == gi) v = -__builtin_inff();  // mask self-attention
        sv[c] = v;
        mx = fmaxf(mx, v);
      }
      const float corr = __expf(m_run - mx);
      float sum = 0.0f;
#pragma unroll
      for (int c = 0; c < 32; ++c) {
        float p = __expf(sv[c] - mx);
        sum += p;
        myP[row * 32 + c] = (_Float16)p;
      }
      l_run = l_run * corr + sum;
      m_run = mx;
      myF[row] = corr;
    }
    asm volatile("s_wait_dscnt 0" ::: "memory");

    // ---- rescale accumulators by the per-row correction factor
    float f[8];
#pragma unroll
    for (int r = 0; r < 8; ++r) f[r] = myF[rbase + r];
#pragma unroll
    for (int t = 0; t < 4; ++t)
#pragma unroll
      for (int r = 0; r < 8; ++r) o[t][r] *= f[r];

    // ---- PV: P(16x32) * V(32x64) via 4 WMMAs (V fragments already resident)
    const v16h pa = load_frag_a(myP, 32);
#pragma unroll
    for (int t = 0; t < 4; ++t) o[t] = wmma_f16(pa, vf[t], o[t]);
  }

  // ---- final 1/l normalization and store (natural [b, n, h*d] layout, f16)
  if (lane < 16) myF[lane] = 1.0f / l_run;
  asm volatile("s_wait_dscnt 0" ::: "memory");
  float f[8];
#pragma unroll
  for (int r = 0; r < 8; ++r) f[r] = myF[rbase + r];
  _Float16* outp = Oh + ((size_t)(b * SEQ) + q0) * DIMC + h * DHEAD;
#pragma unroll
  for (int t = 0; t < 4; ++t)
#pragma unroll
    for (int r = 0; r < 8; ++r)
      outp[(size_t)(rbase + r) * DIMC + t * 16 + colL] =
          (_Float16)(o[t][r] * f[r]);
}

// ---------------------------------------------------------------- launch
extern "C" void kernel_launch(void* const* d_in, const int* in_sizes, int n_in,
                              void* d_out, int out_size, void* d_ws,
                              size_t ws_size, hipStream_t stream) {
  const float* x    = (const float*)d_in[0];
  const float* Wq   = (const float*)d_in[1];
  const float* bq   = (const float*)d_in[2];
  const float* Wk   = (const float*)d_in[3];
  const float* bk   = (const float*)d_in[4];
  const float* Wv   = (const float*)d_in[5];
  const float* bv   = (const float*)d_in[6];
  const float* Wo   = (const float*)d_in[7];
  const float* bo   = (const float*)d_in[8];
  const float* logt = (const float*)d_in[9];
  float* out = (float*)d_out;

  // workspace carve-up (all f16 intermediates; ~92 MB total, L2-resident)
  char* ws = (char*)d_ws;
  size_t off = 0;
  auto take = [&](size_t bytes) {
    void* p = ws + off;
    off += (bytes + 255) & ~(size_t)255;
    return p;
  };
  const size_t tokBytes = (size_t)MROWS * DIMC * sizeof(_Float16);  // 16.8 MB
  const size_t wBytes   = (size_t)DIMC * DIMC * sizeof(_Float16);   // 2.1 MB
  _Float16* xh  = (_Float16*)take(tokBytes);
  _Float16* wqt = (_Float16*)take(wBytes);
  _Float16* wkt = (_Float16*)take(wBytes);
  _Float16* wvt = (_Float16*)take(wBytes);
  _Float16* wot = (_Float16*)take(wBytes);
  _Float16* qh  = (_Float16*)take(tokBytes);
  _Float16* kh  = (_Float16*)take(tokBytes);
  _Float16* vt  = (_Float16*)take(tokBytes);  // [b,h,d,n]
  _Float16* aoh = (_Float16*)take(tokBytes);

  // 1) convert activations + transpose-convert weights to f16
  const int n4 = (MROWS * DIMC) / 4;  // 2,097,152 float4's
  cvt_f32_to_f16<<<(n4 + 255) / 256, 256, 0, stream>>>((const float4*)x,
                                                       (v4h*)xh, n4);
  dim3 tg(DIMC / 16, DIMC / 16);
  transpose_cvt<<<tg, 256, 0, stream>>>(Wq, wqt);
  transpose_cvt<<<tg, 256, 0, stream>>>(Wk, wkt);
  transpose_cvt<<<tg, 256, 0, stream>>>(Wv, wvt);
  transpose_cvt<<<tg, 256, 0, stream>>>(Wo, wot);

  // 2) Q/K/V projections (V written transposed per-head)
  dim3 gg(MROWS / 128, DIMC / 128);
  gemm_wmma<0><<<gg, 256, 0, stream>>>(xh, wqt, bq, qh);
  gemm_wmma<0><<<gg, 256, 0, stream>>>(xh, wkt, bk, kh);
  gemm_wmma<1><<<gg, 256, 0, stream>>>(xh, wvt, bv, vt);

  // 3) masked attention with online softmax
  dim3 ag(SEQ / 128, NHEADS, BATCH);
  attn_kernel<<<ag, 256, 0, stream>>>(qh, kh, vt, logt, aoh);

  // 4) output projection -> f32 d_out
  gemm_wmma<2><<<gg, 256, 0, stream>>>(aoh, wot, bo, (void*)out);
}
